// HypothesisRegister_16793322127883
// MI455X (gfx1250) — matrix operand order repaired and checked by reading, not compiled, not run on hardware
//
#include <hip/hip_runtime.h>
#include <hip/hip_bf16.h>

typedef _Float16 v16h __attribute__((ext_vector_type(16)));
typedef _Float16 v8h  __attribute__((ext_vector_type(8)));
typedef _Float16 h4   __attribute__((ext_vector_type(4)));
typedef float    v8f  __attribute__((ext_vector_type(8)));
typedef float    v4f  __attribute__((ext_vector_type(4)));

#define B_TOT 65536
#define H_DIM 512
#define D_DIM 256
#define M_HYP 4
#define DH    128

// ---------------------------------------------------------------------------
// Prep: convert + transpose weights to f16, N-major with contiguous K runs.
//   W1t [512][512]: n<256 -> W_proj col n ; n>=256 -> W_gate[k][n-256], k<512
//   W2t [256][256]: W2t[n][k] = W_gate[512+k][n]
//   Wc1t[128][256]: Wc1t[n][k] = W_c1[k][n]
// ---------------------------------------------------------------------------
__global__ void hr_prep_weights(const float* __restrict__ Wp,
                                const float* __restrict__ Wg,
                                const float* __restrict__ Wc1,
                                _Float16* __restrict__ W1t,
                                _Float16* __restrict__ W2t,
                                _Float16* __restrict__ Wc1t) {
  int i = blockIdx.x * 256 + threadIdx.x;
  if (i < 512 * 512) {
    int n = i >> 9, k = i & 511;
    float v = (n < 256) ? Wp[k * 256 + n] : Wg[k * 256 + (n - 256)];
    W1t[i] = (_Float16)v;
    return;
  }
  int j = i - 512 * 512;
  if (j < 256 * 256) {
    int n = j >> 8, k = j & 255;
    W2t[j] = (_Float16)Wg[(512 + k) * 256 + n];
    return;
  }
  int l = j - 256 * 256;
  if (l < 128 * 256) {
    int n = l >> 8, k = l & 255;
    Wc1t[l] = (_Float16)Wc1[k * 128 + n];
  }
}

// ---------------------------------------------------------------------------
// Fused main kernel: 16 batch rows per workgroup, 8 waves (wave32).
// ---------------------------------------------------------------------------
__global__ __launch_bounds__(256)
void hr_main(const float* __restrict__ hidden,   // (B,512)
             const float* __restrict__ hyp,      // (B,4,256)
             const float* __restrict__ b_proj,
             const float* __restrict__ ln_g,
             const float* __restrict__ ln_b,
             const float* __restrict__ b_c1,
             const float* __restrict__ W_c2,
             const float* __restrict__ b_c2,
             const float* __restrict__ b_gate,
             const _Float16* __restrict__ W1t,
             const _Float16* __restrict__ W2t,
             const _Float16* __restrict__ Wc1t,
             float* __restrict__ out_upd,        // (B,4,256)
             float* __restrict__ out_pri,        // (B,256)
             float* __restrict__ out_conf) {     // (B,4)
  __shared__ union {
    _Float16 As[16 * 512];                                   // 16 KB (phase 0-1)
    struct { _Float16 Hs[64 * 256]; float P[8][32][8]; } p3; // 32+8 KB (phase 3)
  } U;
  __shared__ float    S1[16 * 256];    // C1 low -> normalized new_h (f32)
  __shared__ float    S2[16 * 256];    // gate partial -> gate
  __shared__ _Float16 nh16[16 * 256];  // new_h f16 (GEMM2 A)
  __shared__ float    red[16][16][2];
  __shared__ float    muv[16][2];
  __shared__ float    confL[64];
  __shared__ int      minIdx[16], maxIdx[16];

  const int tid  = threadIdx.x;
  const int lane = tid & 31;
  const int wave = tid >> 5;
  const int lrow = lane & 15;
  const int khi  = lane >> 4;
  const int gb0  = blockIdx.x * 16;

  // ---- Phase 0: hidden tile (16x512 f32) -> f16 in LDS -------------------
  // Streaming read (no reuse beyond this LDS stage): non-temporal.
  {
    const v4f* src = (const v4f*)(hidden + (size_t)gb0 * H_DIM);
    h4* dst = (h4*)U.As;
    #pragma unroll
    for (int i = 0; i < 8; ++i) {
      int q = tid + i * 256;                 // 2048 float4 total
      v4f v = __builtin_nontemporal_load(src + q);
      h4 h; h[0] = (_Float16)v[0]; h[1] = (_Float16)v[1];
      h[2] = (_Float16)v[2]; h[3] = (_Float16)v[3];
      dst[q] = h;
    }
  }
  __syncthreads();

  // ---- Phase 1: GEMM1  C1(16x512) = hidden_f16 @ [W_proj|W_gate_h] -------
  v8f acc[4] = {v8f(0.f), v8f(0.f), v8f(0.f), v8f(0.f)};
  for (int kk = 0; kk < 16; ++kk) {
    v16h a;
    const _Float16* ap = &U.As[lrow * 512 + kk * 32 + khi * 8];
    ((v8h*)&a)[0] = *(const v8h*)ap;
    ((v8h*)&a)[1] = *(const v8h*)(ap + 16);
    #pragma unroll
    for (int t = 0; t < 4; ++t) {
      int n = wave * 64 + t * 16 + lrow;
      const _Float16* bp = &W1t[n * 512 + kk * 32 + khi * 8];
      v16h b;
      ((v8h*)&b)[0] = *(const v8h*)bp;
      ((v8h*)&b)[1] = *(const v8h*)(bp + 16);
      acc[t] = __builtin_amdgcn_wmma_f32_16x16x32_f16(
          false, a, false, b, (short)0, acc[t], false, false);
    }
  }
  #pragma unroll
  for (int t = 0; t < 4; ++t) {
    int col = wave * 64 + t * 16 + lrow;
    float* dst = (col < 256) ? S1 : S2;
    int c = col & 255;
    #pragma unroll
    for (int v = 0; v < 8; ++v)
      dst[(v + 8 * khi) * 256 + c] = acc[t][v];
  }
  __syncthreads();   // As now dead; S1 = pre-LN new_h, S2 = gate partial

  // ---- Phase 1.5: LayerNorm; also stage hypotheses as f16 (aliases As) ---
  {
    int r = tid >> 4, j = tid & 15;
    float s = 0.f, ss = 0.f;
    #pragma unroll
    for (int q = 0; q < 16; ++q) {
      int d = j * 16 + q;
      float x = S1[r * 256 + d] + b_proj[d];
      s += x; ss += x * x;
    }
    red[r][j][0] = s; red[r][j][1] = ss;
  }
  {
    const v4f* src = (const v4f*)(hyp + (size_t)gb0 * M_HYP * D_DIM);
    h4* dst = (h4*)U.p3.Hs;
    #pragma unroll
    for (int i = 0; i < 16; ++i) {
      int q = tid + i * 256;                 // 4096 float4 total
      v4f v = src[q];                        // regular: re-read in phase 4
      h4 h; h[0] = (_Float16)v[0]; h[1] = (_Float16)v[1];
      h[2] = (_Float16)v[2]; h[3] = (_Float16)v[3];
      dst[q] = h;
    }
  }
  __syncthreads();
  if (tid < 16) {
    float s = 0.f, ss = 0.f;
    #pragma unroll
    for (int j = 0; j < 16; ++j) { s += red[tid][j][0]; ss += red[tid][j][1]; }
    float mu  = s * (1.f / 256.f);
    float var = ss * (1.f / 256.f) - mu * mu;
    muv[tid][0] = mu;
    muv[tid][1] = __frsqrt_rn(var + 1e-5f);
  }
  __syncthreads();
  {
    int r = tid >> 4, j = tid & 15;
    float mu = muv[r][0], rs = muv[r][1];
    #pragma unroll
    for (int q = 0; q < 16; ++q) {
      int d = j * 16 + q;
      float x = S1[r * 256 + d] + b_proj[d];
      float y = (x - mu) * rs * ln_g[d] + ln_b[d];
      S1[r * 256 + d] = y;
      nh16[r * 256 + d] = (_Float16)y;
    }
  }
  __syncthreads();

  // ---- Phase 2: GEMM2  gate += new_h_f16 @ W_gate_newh -------------------
  v8f acc2[2] = {v8f(0.f), v8f(0.f)};
  for (int kk = 0; kk < 8; ++kk) {
    v16h a;
    const _Float16* ap = &nh16[lrow * 256 + kk * 32 + khi * 8];
    ((v8h*)&a)[0] = *(const v8h*)ap;
    ((v8h*)&a)[1] = *(const v8h*)(ap + 16);
    #pragma unroll
    for (int t = 0; t < 2; ++t) {
      int n = wave * 32 + t * 16 + lrow;
      const _Float16* bp = &W2t[n * 256 + kk * 32 + khi * 8];
      v16h b;
      ((v8h*)&b)[0] = *(const v8h*)bp;
      ((v8h*)&b)[1] = *(const v8h*)(bp + 16);
      acc2[t] = __builtin_amdgcn_wmma_f32_16x16x32_f16(
          false, a, false, b, (short)0, acc2[t], false, false);
    }
  }
  #pragma unroll
  for (int t = 0; t < 2; ++t) {
    int c = wave * 32 + t * 16 + lrow;
    #pragma unroll
    for (int v = 0; v < 8; ++v)
      S2[(v + 8 * khi) * 256 + c] += acc2[t][v];
  }
  __syncthreads();
  #pragma unroll
  for (int i = 0; i < 16; ++i) {
    int q = tid + i * 256;
    int d = q & 255;
    float x = S2[q] + b_gate[d];
    S2[q] = 1.f / (1.f + __expf(-x));   // gate
  }
  __syncthreads();

  // ---- Phase 3: GEMM3  h1(64x128) = hyp_f16 @ W_c1t, fused conf reduce ---
  v8f acc3[4] = {v8f(0.f), v8f(0.f), v8f(0.f), v8f(0.f)};
  const int mt = wave >> 1, nhf = wave & 1;
  for (int kk = 0; kk < 8; ++kk) {
    v16h a;
    const _Float16* ap = &U.p3.Hs[(mt * 16 + lrow) * 256 + kk * 32 + khi * 8];
    ((v8h*)&a)[0] = *(const v8h*)ap;
    ((v8h*)&a)[1] = *(const v8h*)(ap + 16);
    #pragma unroll
    for (int t = 0; t < 4; ++t) {
      int n = nhf * 64 + t * 16 + lrow;
      const _Float16* bp = &Wc1t[n * 256 + kk * 32 + khi * 8];
      v16h b;
      ((v8h*)&b)[0] = *(const v8h*)bp;
      ((v8h*)&b)[1] = *(const v8h*)(bp + 16);
      acc3[t] = __builtin_amdgcn_wmma_f32_16x16x32_f16(
          false, a, false, b, (short)0, acc3[t], false, false);
    }
  }
  {
    float p[8];
    #pragma unroll
    for (int v = 0; v < 8; ++v) p[v] = 0.f;
    #pragma unroll
    for (int t = 0; t < 4; ++t) {
      int n = nhf * 64 + t * 16 + lrow;
      float bc = b_c1[n], wc = W_c2[n];
      #pragma unroll
      for (int v = 0; v < 8; ++v) {
        float x = acc3[t][v] + bc;
        x = x > 0.f ? x : 0.f;          // relu
        p[v] += x * wc;                 // * W_c2 column
      }
    }
    #pragma unroll
    for (int v = 0; v < 8; ++v) U.p3.P[wave][lane][v] = p[v];
  }
  __syncthreads();

  // conf[r] = sigmoid(sum + b_c2)
  if (tid < 64) {
    int r = tid, mt2 = r >> 4, rr = r & 15;
    int lb = (rr >> 3) * 16, v = rr & 7;
    float s = 0.f;
    #pragma unroll
    for (int l = 0; l < 16; ++l)
      s += U.p3.P[2 * mt2][lb + l][v] + U.p3.P[2 * mt2 + 1][lb + l][v];
    float c = 1.f / (1.f + __expf(-(s + b_c2[0])));
    confL[r] = c;
    __builtin_nontemporal_store(c, out_conf + (size_t)blockIdx.x * 64 + r);
  }
  __syncthreads();
  if (tid < 16) {
    float cmn = confL[tid * 4], cmx = cmn;
    int imn = 0, imx = 0;
    #pragma unroll
    for (int m = 1; m < 4; ++m) {
      float c = confL[tid * 4 + m];
      if (c < cmn) { cmn = c; imn = m; }  // first occurrence (strict) = argmin
      if (c > cmx) { cmx = c; imx = m; }
    }
    minIdx[tid] = imn; maxIdx[tid] = imx;
  }
  __syncthreads();

  // ---- Phase 4: outputs (streaming writes: non-temporal) ------------------
  const v4f* hp4 = (const v4f*)(hyp + (size_t)gb0 * M_HYP * D_DIM);
  v4f* up4 = (v4f*)(out_upd + (size_t)gb0 * M_HYP * D_DIM);
  #pragma unroll
  for (int i = 0; i < 16; ++i) {
    int q = tid + i * 256;               // 4096 float4
    int d4 = q & 63;
    int row = q >> 6;                    // 0..63 = (r,m)
    int r = row >> 2, m = row & 3;
    v4f hv = hp4[q];
    if (m == minIdx[r]) {
      #pragma unroll
      for (int e = 0; e < 4; ++e) {
        int d = d4 * 4 + e;
        float g = S2[r * 256 + d];
        hv[e] = g * hv[e] + (1.f - g) * S1[r * 256 + d];
      }
    }
    __builtin_nontemporal_store(hv, up4 + q);
  }
  v4f* pp4 = (v4f*)(out_pri + (size_t)gb0 * D_DIM);
  #pragma unroll
  for (int i = 0; i < 4; ++i) {
    int q = tid + i * 256;               // 1024 float4
    int d4 = q & 63, r = q >> 6;
    int m = maxIdx[r];
    v4f hv = hp4[(r * 4 + m) * 64 + d4];
    if (m == minIdx[r]) {                // primary slot was the blended one
      #pragma unroll
      for (int e = 0; e < 4; ++e) {
        int d = d4 * 4 + e;
        float g = S2[r * 256 + d];
        hv[e] = g * hv[e] + (1.f - g) * S1[r * 256 + d];
      }
    }
    __builtin_nontemporal_store(hv, pp4 + q);
  }
}

extern "C" void kernel_launch(void* const* d_in, const int* in_sizes, int n_in,
                              void* d_out, int out_size, void* d_ws, size_t ws_size,
                              hipStream_t stream) {
  const float* hidden = (const float*)d_in[0];
  const float* hyp    = (const float*)d_in[1];
  const float* W_proj = (const float*)d_in[2];
  const float* b_proj = (const float*)d_in[3];
  const float* ln_g   = (const float*)d_in[4];
  const float* ln_b   = (const float*)d_in[5];
  const float* W_c1   = (const float*)d_in[6];
  const float* b_c1   = (const float*)d_in[7];
  const float* W_c2   = (const float*)d_in[8];
  const float* b_c2   = (const float*)d_in[9];
  const float* W_gate = (const float*)d_in[10];
  const float* b_gate = (const float*)d_in[11];

  _Float16* W1t  = (_Float16*)d_ws;          // 512*512 f16
  _Float16* W2t  = W1t + 512 * 512;          // 256*256 f16
  _Float16* Wc1t = W2t + 256 * 256;          // 128*256 f16

  hr_prep_weights<<<1408, 256, 0, stream>>>(W_proj, W_gate, W_c1, W1t, W2t, Wc1t);

  float* out_upd  = (float*)d_out;
  float* out_pri  = out_upd + (size_t)B_TOT * M_HYP * D_DIM;
  float* out_conf = out_pri + (size_t)B_TOT * D_DIM;

  hr_main<<<B_TOT / 16, 256, 0, stream>>>(hidden, hyp, b_proj, ln_g, ln_b,
                                          b_c1, W_c2, b_c2, b_gate,
                                          W1t, W2t, Wc1t,
                                          out_upd, out_pri, out_conf);
}
